// NoteAxis_60662118088768
// MI455X (gfx1250) — matrix operand order repaired and checked by reading, not compiled, not run on hardware
//
#include <hip/hip_runtime.h>
#include <hip/hip_bf16.h>

// ---------------------------------------------------------------------------
// 2-layer LSTM (U=128) over N=128 steps, B=4096, via bf16 WMMA on gfx1250.
// ---------------------------------------------------------------------------

typedef __attribute__((ext_vector_type(16))) __bf16 v16bf;
typedef __attribute__((ext_vector_type(8)))  float  v8f;

union Frag16 {
    v16bf v;
    uint4 q[2];
    unsigned short u[16];
};

__device__ __forceinline__ unsigned short f2bf(float f) {
    unsigned int x = __float_as_uint(f);
    unsigned int r = x + 0x7FFFu + ((x >> 16) & 1u);   // round-to-nearest-even
    return (unsigned short)(r >> 16);
}

__device__ __forceinline__ float bf2f(unsigned short u) {
    return __uint_as_float(((unsigned int)u) << 16);
}

__device__ __forceinline__ float sigm(float x) {
    return 1.0f / (1.0f + __expf(-x));
}

// LICM barrier: an opaque uniform zero offset. Added to a kernarg-rooted
// pointer it keeps global address-space provenance (loads lower to
// global_load_b128, LOADcnt only) while making the address non-invariant in
// the timestep loop, so ~512 VGPRs of weight fragments are never hoisted
// and spilled to scratch.
__device__ __forceinline__ unsigned opaque_zero() {
    unsigned v = 0;
    asm volatile("" : "+s"(v));
    return v;
}

// A-fragment: 16x32 bf16 tile (M x K) from LDS buffer laid out [16][128].
// ISA layout: lanes 0-15 -> M=lane, K base 0; lanes 16-31 -> M=lane-16, K base +8.
// Vector elems 0..7  = K (ks + kb .. ks + kb + 7)
// Vector elems 8..15 = K (ks + 16 + kb .. ks + 16 + kb + 7)
__device__ __forceinline__ Frag16 loadA(const unsigned short* __restrict__ h,
                                        int lane, int ks) {
    Frag16 a;
    int m  = lane & 15;
    int kb = (lane < 16) ? 0 : 8;
    a.q[0] = *(const uint4*)(h + m * 128 + ks + kb);
    a.q[1] = *(const uint4*)(h + m * 128 + ks + 16 + kb);
    return a;
}

// B-fragment: 32x16 bf16 tile (K x N) where B = W^T, W is [512][128] row-major
// bf16 in global. Lane = output column (gate unit g); each lane holds K in the
// same order as loadA: elems 0..7 = K ks+kb..+7, elems 8..15 = K ks+16+kb..+7 (kb = 0/8).
__device__ __forceinline__ Frag16 loadB(const unsigned short* __restrict__ w,
                                        int lane, int col0, int ks) {
    Frag16 b;
    int g  = col0 + (lane & 15);
    int kb = ks + ((lane < 16) ? 0 : 8);
    const uint4* p = (const uint4*)(w + g * 128 + kb);
    b.q[0] = p[0];
    b.q[1] = p[2];
    return b;
}

// C/D fp32 16x16: VGPR r holds row r (lanes 0-15) / row r+8 (lanes 16-31),
// col = lane & 15.
__device__ __forceinline__ void storeC(float* __restrict__ gates, v8f acc,
                                       int lane, int col0) {
    int col = col0 + (lane & 15);
    int rb  = (lane < 16) ? 0 : 8;
#pragma unroll
    for (int r = 0; r < 8; ++r)
        gates[(rb + r) * 512 + col] = acc[r];
}

// gates[16][512] = A0[16][128] @ W0^T + A1[16][128] @ W1^T
// (each wave produces its own 16x64 column slice)
__device__ __forceinline__ void dual_mm(const unsigned short* __restrict__ a0lds,
                                        const unsigned short* __restrict__ w0_in,
                                        const unsigned short* __restrict__ a1lds,
                                        const unsigned short* __restrict__ w1_in,
                                        float* __restrict__ gates,
                                        int wave, int lane) {
    // Re-materialize weight addresses each call so weight loads stay in-loop.
    const unsigned short* w0 = w0_in + opaque_zero();
    const unsigned short* w1 = w1_in + opaque_zero();

    Frag16 a0[4], a1[4];
#pragma unroll
    for (int k = 0; k < 4; ++k) {
        a0[k] = loadA(a0lds, lane, k * 32);
        a1[k] = loadA(a1lds, lane, k * 32);
    }
#pragma unroll
    for (int ct = 0; ct < 4; ++ct) {
        int col0 = wave * 64 + ct * 16;
        v8f acc = {};
#pragma unroll
        for (int k = 0; k < 4; ++k) {
            Frag16 b = loadB(w0, lane, col0, k * 32);
            acc = __builtin_amdgcn_wmma_f32_16x16x32_bf16(
                false, a0[k].v, false, b.v, (short)0, acc, false, false);
        }
#pragma unroll
        for (int k = 0; k < 4; ++k) {
            Frag16 b = loadB(w1, lane, col0, k * 32);
            acc = __builtin_amdgcn_wmma_f32_16x16x32_bf16(
                false, a1[k].v, false, b.v, (short)0, acc, false, false);
        }
        storeC(gates, acc, lane, col0);
    }
}

// LSTM pointwise stage: gates[16][512] (+bias) with c -> new c (fp32 LDS) and
// new h (bf16 LDS). Gate order i, f, g, o.
__device__ __forceinline__ void lstm_act(const float* __restrict__ gates,
                                         const float* __restrict__ bias,
                                         float* __restrict__ cbuf,
                                         unsigned short* __restrict__ hbuf,
                                         int tid) {
#pragma unroll
    for (int it = 0; it < 8; ++it) {
        int idx = tid + it * 256;             // 0..2047
        int m = idx >> 7;
        int u = idx & 127;
        float ig = gates[m * 512 + u]        + bias[u];
        float fg = gates[m * 512 + 128 + u]  + bias[128 + u];
        float gg = gates[m * 512 + 256 + u]  + bias[256 + u];
        float og = gates[m * 512 + 384 + u]  + bias[384 + u];
        float c  = sigm(fg) * cbuf[idx] + sigm(ig) * tanhf(gg);
        cbuf[idx] = c;
        hbuf[idx] = f2bf(sigm(og) * tanhf(c));
    }
}

// ---------------------------------------------------------------------------
// Prep: fp32 -> bf16 weights, combined biases.
// ---------------------------------------------------------------------------
__global__ void noteaxis_prep_kernel(const float* __restrict__ wih0,
                                     const float* __restrict__ whh0,
                                     const float* __restrict__ wih1,
                                     const float* __restrict__ whh1,
                                     const float* __restrict__ bih0,
                                     const float* __restrict__ bhh0,
                                     const float* __restrict__ bih1,
                                     const float* __restrict__ bhh1,
                                     unsigned short* __restrict__ wb,
                                     float* __restrict__ bias0,
                                     float* __restrict__ bias1) {
    int idx = blockIdx.x * 256 + threadIdx.x;
    if (idx < 4 * 65536) {
        int m = idx >> 16;
        int r = idx & 65535;
        const float* src = (m == 0) ? wih0 : (m == 1) ? whh0 : (m == 2) ? wih1 : whh1;
        wb[idx] = f2bf(src[r]);
    }
    if (idx < 512) {
        bias0[idx] = bih0[idx] + bhh0[idx];
    } else if (idx < 1024) {
        int r = idx - 512;
        bias1[r] = bih1[r] + bhh1[r];
    }
}

// ---------------------------------------------------------------------------
// Main recurrent kernel: one block = 16 batch rows, 8 wave32s, 128 timesteps.
// ---------------------------------------------------------------------------
__global__ void __launch_bounds__(256)
noteaxis_lstm_kernel(const float* __restrict__ nf,      // [4096][128][127]
                     const float* __restrict__ tgt,     // [4096][128]
                     const unsigned short* __restrict__ wih0,
                     const unsigned short* __restrict__ whh0,
                     const unsigned short* __restrict__ wih1,
                     const unsigned short* __restrict__ whh1,
                     const float* __restrict__ bias0,   // [512]
                     const float* __restrict__ bias1,   // [512]
                     const float* __restrict__ w_out,   // [128]
                     const float* __restrict__ b_out,   // [1]
                     float* __restrict__ out) {         // [4096][128]
    __shared__ unsigned short xbuf[16 * 128];
    __shared__ unsigned short h0buf[16 * 128];
    __shared__ unsigned short h1buf[16 * 128];
    __shared__ float c0buf[16 * 128];
    __shared__ float c1buf[16 * 128];
    __shared__ float gates[16 * 512];

    const int tid  = threadIdx.x;
    const int lane = tid & 31;
    const int wave = tid >> 5;
    const int b0   = blockIdx.x * 16;

#pragma unroll
    for (int it = 0; it < 8; ++it) {
        int i = tid + it * 256;
        h0buf[i] = 0;
        h1buf[i] = 0;
        c0buf[i] = 0.0f;
        c1buf[i] = 0.0f;
    }
    __syncthreads();

    const float bo = b_out[0];

#pragma unroll 1
    for (int n = 0; n < 128; ++n) {
        // Stage x tile: [16][128] = 127 note features + teacher-forced cond.
#pragma unroll
        for (int it = 0; it < 8; ++it) {
            int i = tid + it * 256;
            int m = i >> 7;
            int j = i & 127;
            float v;
            if (j < 127)
                v = nf[(size_t)(b0 + m) * (128 * 127) + (size_t)n * 127 + j];
            else
                v = (n == 0) ? 0.0f : tgt[(size_t)(b0 + m) * 128 + (n - 1)];
            xbuf[i] = f2bf(v);
        }
        __syncthreads();

        // Layer 0: gates = x @ Wih0^T + h0 @ Whh0^T
        dual_mm(xbuf, wih0, h0buf, whh0, gates, wave, lane);
        __syncthreads();
        lstm_act(gates, bias0, c0buf, h0buf, tid);
        __syncthreads();

        // Layer 1: gates = h0 @ Wih1^T + h1 @ Whh1^T
        dual_mm(h0buf, wih1, h1buf, whh1, gates, wave, lane);
        __syncthreads();
        lstm_act(gates, bias1, c1buf, h1buf, tid);
        __syncthreads();

        // Output head: sigmoid(h1 . w_out + b_out)
        if (tid < 16) {
            float s = 0.0f;
#pragma unroll 4
            for (int u = 0; u < 128; ++u)
                s += bf2f(h1buf[tid * 128 + u]) * w_out[u];
            out[(size_t)(b0 + tid) * 128 + n] = sigm(s + bo);
        }
        __syncthreads();
    }
}

extern "C" void kernel_launch(void* const* d_in, const int* in_sizes, int n_in,
                              void* d_out, int out_size, void* d_ws, size_t ws_size,
                              hipStream_t stream) {
    (void)in_sizes; (void)n_in; (void)out_size; (void)ws_size;

    const float* nf   = (const float*)d_in[0];
    const float* tgt  = (const float*)d_in[1];
    const float* wih0 = (const float*)d_in[2];
    const float* whh0 = (const float*)d_in[3];
    const float* bih0 = (const float*)d_in[4];
    const float* bhh0 = (const float*)d_in[5];
    const float* wih1 = (const float*)d_in[6];
    const float* whh1 = (const float*)d_in[7];
    const float* bih1 = (const float*)d_in[8];
    const float* bhh1 = (const float*)d_in[9];
    const float* wout = (const float*)d_in[10];
    const float* bout = (const float*)d_in[11];
    float* out = (float*)d_out;

    // Workspace layout: 4x [512x128] bf16 weights, then 2x [512] fp32 biases.
    char* ws = (char*)d_ws;
    unsigned short* wb  = (unsigned short*)ws;           // 4 * 65536 ushort = 512 KB
    float* bias0 = (float*)(ws + 4u * 65536u * 2u);      // 2 KB
    float* bias1 = (float*)(ws + 4u * 65536u * 2u + 2048u);

    noteaxis_prep_kernel<<<1024, 256, 0, stream>>>(
        wih0, whh0, wih1, whh1, bih0, bhh0, bih1, bhh1, wb, bias0, bias1);

    noteaxis_lstm_kernel<<<256, 256, 0, stream>>>(
        nf, tgt,
        wb, wb + 65536, wb + 2 * 65536, wb + 3 * 65536,
        bias0, bias1, wout, bout, out);
}
